// depthNet_7773890806199
// MI455X (gfx1250) — compile-verified, matrix operand order checked by reading, not run on hardware
//
#include <hip/hip_runtime.h>
#include <hip/hip_bf16.h>
#include <math.h>

// ---------------------------------------------------------------------------
// CDNA5 WMMA (wave32): D(16x16 f32) = A(16x32 f16) x B(32x16 f16) + C
// ---------------------------------------------------------------------------
typedef __attribute__((ext_vector_type(16))) _Float16 v16h;
typedef __attribute__((ext_vector_type(8)))  _Float16 v8h;
typedef __attribute__((ext_vector_type(8)))  float    v8f;
typedef __attribute__((ext_vector_type(4)))  int      v4i;
typedef __attribute__((address_space(1))) v4i gv4i;   // global int4
typedef __attribute__((address_space(3))) v4i lv4i;   // LDS int4

union H8 { v8h v; unsigned u[4]; };

#define BN_EPS 1e-5f

// ===========================================================================
// Cost volume -> network input X, NHWC fp16, channel stride 96 (67 real + pad0)
// ===========================================================================
__device__ __forceinline__ float samp_zero(const float* img, int x, int y, int W, int H) {
    if (x < 0 || x > W - 1 || y < 0 || y > H - 1) return 0.0f;
    return img[(size_t)y * W + x];
}

__global__ void cost_volume_kernel(const float* __restrict__ left,
                                   const float* __restrict__ right,
                                   const float* __restrict__ krkiuv,
                                   const float* __restrict__ kt,
                                   _Float16* __restrict__ X,   // NHWC, stride 96
                                   int B, int H, int W)
{
    int idx = blockIdx.x * blockDim.x + threadIdx.x;
    int HW = H * W;
    if (idx >= B * HW) return;
    int b = idx / HW, r = idx % HW;
    int h = r / W, w = r % W;
    int lin = w * H + h;   // reference flattens (W,H) u-major

    const float* K0 = krkiuv + (size_t)b * 3 * HW;
    const float* T0 = kt     + (size_t)b * 3 * HW;
    float k0 = K0[lin], k1 = K0[HW + lin], k2 = K0[2 * HW + lin];
    float t0 = T0[lin], t1 = T0[HW + lin], t2 = T0[2 * HW + lin];

    float l0 = left[((size_t)b * 3 + 0) * HW + r];
    float l1 = left[((size_t)b * 3 + 1) * HW + r];
    float l2 = left[((size_t)b * 3 + 2) * HW + r];

    _Float16* xp = X + ((size_t)b * HW + r) * 96;
    xp[0] = (_Float16)l0;
    xp[1] = (_Float16)l1;
    xp[2] = (_Float16)l2;

    const float idb = 1.0f / 50.0f;
    const float ids = (2.0f - 1.0f / 50.0f) / 63.0f;
    const float* R0 = right + (size_t)b * 3 * HW;
    float halfW = 0.5f * (float)W, halfH = 0.5f * (float)H;

    for (int d = 0; d < 64; ++d) {
        float depth = 1.0f / (idb + (float)d * ids);
        float tx = k0 * depth + t0;
        float ty = k1 * depth + t1;
        float tz = k2 * depth + t2;
        float inv = 1.0f / (tz + 1e-6f);
        float u = tx * inv, v = ty * inv;
        float gx = (u - halfW) / halfW;
        float gy = (v - halfH) / halfH;
        float xx = (gx + 1.0f) * 0.5f * (float)(W - 1);
        float yy = (gy + 1.0f) * 0.5f * (float)(H - 1);
        float x0f = floorf(xx), y0f = floorf(yy);
        float wx = xx - x0f, wy = yy - y0f;
        int x0 = (int)x0f, y0 = (int)y0f;

        float cost = 0.0f;
        #pragma unroll
        for (int c = 0; c < 3; ++c) {
            const float* img = R0 + (size_t)c * HW;
            float v00 = samp_zero(img, x0,     y0,     W, H);
            float v01 = samp_zero(img, x0 + 1, y0,     W, H);
            float v10 = samp_zero(img, x0,     y0 + 1, W, H);
            float v11 = samp_zero(img, x0 + 1, y0 + 1, W, H);
            float val = v00 * (1.f - wx) * (1.f - wy) + v01 * wx * (1.f - wy)
                      + v10 * (1.f - wx) * wy         + v11 * wx * wy;
            float lc = (c == 0) ? l0 : ((c == 1) ? l1 : l2);
            cost += fabsf(val - lc);
        }
        xp[3 + d] = (_Float16)cost;
    }
    #pragma unroll
    for (int c = 67; c < 96; ++c) xp[c] = (_Float16)0.f;   // zero pad channels
}

// ===========================================================================
// Weight repack: OIHW f32 -> fragment-major fp16
//   layout [ky][kx][ntile][kchunk][lane(32)][elem(16)]
//   elem j of lane: n = ntile*16 + (lane&15); hi = lane>>4;
//                   ci = kchunk*32 + 16*(j>>3) + 8*hi + (j&7)
// so each lane's B fragment is 16 contiguous halfs (32B).
// ===========================================================================
__global__ void repack_weight_kernel(const float* __restrict__ w,  // [Cout][Cin][Kk][Kk]
                                     _Float16* __restrict__ o,
                                     int Cout, int Cin, int Kk, int ntiles, int nchunks)
{
    size_t total = (size_t)Kk * Kk * ntiles * nchunks * 512;
    size_t idx = (size_t)blockIdx.x * blockDim.x + threadIdx.x;
    if (idx >= total) return;
    int j    = (int)(idx & 15);
    int lane = (int)((idx >> 4) & 31);
    size_t t = idx >> 9;
    int kc  = (int)(t % nchunks); t /= nchunks;
    int nt  = (int)(t % ntiles);  t /= ntiles;
    int kxy = (int)t;
    int ky = kxy / Kk, kx = kxy - ky * Kk;
    int n  = nt * 16 + (lane & 15);
    int hi = lane >> 4;
    int ci = kc * 32 + 16 * (j >> 3) + 8 * hi + (j & 7);
    float v = 0.0f;
    if (n < Cout && ci < Cin)
        v = w[(((size_t)n * Cin + ci) * Kk + ky) * Kk + kx];
    o[idx] = (_Float16)v;
}

// ===========================================================================
// Implicit-GEMM conv, NHWC fp16 activations, fragment-major fp16 weights.
// Block = 128 threads = 4 waves. Each wave owns a 16(M=pixels) x 32(N=cout)
// tile (two adjacent 16-wide n-tiles -> one A load feeds two WMMAs).
// All waves share the block's n-tile pair: B slab (2*nchunks*1KB <= 64KB)
// staged in LDS per (ky,kx) via async global->LDS DMA when available.
// Fused BN scale+bias+ReLU epilogue; channel-offset views handle concats.
// ===========================================================================
__global__ __launch_bounds__(128)
void conv_wmma_kernel(const _Float16* __restrict__ in,    // NHWC, stride CinStride
                      const _Float16* __restrict__ wgt,   // fragment-major
                      const float* __restrict__ gamma,
                      const float* __restrict__ beta,
                      _Float16* __restrict__ out,         // NHWC, stride CdstStride
                      int Cin, int CinStride, int CinOff, int Hin, int Win,
                      int Cout, int Ho, int Wo, int CdstStride, int CoffDst,
                      int Kk, int stride, int pad)
{
    __shared__ __align__(32) _Float16 smemB[32768];   // 64KB: 2 tiles * 32 chunks * 512

    int b    = blockIdx.z;
    int tid  = threadIdx.x;
    int lane = tid & 31;
    int wv   = tid >> 5;
    int l15  = lane & 15;
    int hi   = lane >> 4;

    int M       = Ho * Wo;
    int mbase   = (blockIdx.x * 4 + wv) * 16;
    int ntPair  = blockIdx.y;              // n-tiles 2*ntPair, 2*ntPair+1
    int ntiles  = gridDim.y * 2;
    int nchunks = (Cin + 31) >> 5;

    // A-row (output pixel) coordinates for this lane
    int  m     = mbase + l15;
    bool mval  = (m < M);
    int  msafe = mval ? m : 0;
    int  oy = msafe / Wo;
    int  ox = msafe - oy * Wo;
    int  y0 = oy * stride - pad;
    int  x0 = ox * stride - pad;

    const _Float16* inb = in + (size_t)b * Hin * Win * CinStride;
    int aoff = CinOff + 8 * hi;

    v8f acc0 = {0.f, 0.f, 0.f, 0.f, 0.f, 0.f, 0.f, 0.f};
    v8f acc1 = {0.f, 0.f, 0.f, 0.f, 0.f, 0.f, 0.f, 0.f};
    int slabWords = nchunks * 128;   // 16B words for the 2-tile slab

    for (int ky = 0; ky < Kk; ++ky) {
        for (int kx = 0; kx < Kk; ++kx) {
            int kxy = ky * Kk + kx;
            const v4i* wblk =
                (const v4i*)(wgt + ((size_t)(kxy * ntiles + 2 * ntPair) * nchunks) * 512);
            __syncthreads();                     // previous slab fully consumed
#if defined(__gfx1250__) && __has_builtin(__builtin_amdgcn_global_load_async_to_lds_b128)
            for (int i = tid; i < slabWords; i += 128) {
                __builtin_amdgcn_global_load_async_to_lds_b128(
                    (gv4i*)(wblk + i),
                    (lv4i*)(((v4i*)smemB) + i),
                    0, 0);
            }
#if __has_builtin(__builtin_amdgcn_s_wait_asynccnt)
            __builtin_amdgcn_s_wait_asynccnt(0);
#else
            asm volatile("s_wait_asynccnt 0" ::: "memory");
#endif
#else
            for (int i = tid; i < slabWords; i += 128)
                ((v4i*)smemB)[i] = wblk[i];
#endif
            __syncthreads();                     // slab visible to all waves
            // hint-prefetch next slab while we compute on this one
            __builtin_prefetch((const void*)(wblk + slabWords), 0, 1);

            int y = y0 + ky, x = x0 + kx;
            bool rv = mval && (y >= 0) && (y < Hin) && (x >= 0) && (x < Win);
            int ys = rv ? y : 0, xs = rv ? x : 0;
            const _Float16* abase = inb + ((size_t)ys * Win + xs) * CinStride + aoff;

            for (int kc = 0; kc < nchunks; ++kc) {
                H8 lo, hi8;
                lo.u[0] = lo.u[1] = lo.u[2] = lo.u[3] = 0u;
                hi8.u[0] = hi8.u[1] = hi8.u[2] = hi8.u[3] = 0u;
                if (rv) {
                    lo.v  = *(const v8h*)(abase + kc * 32);
                    hi8.v = *(const v8h*)(abase + kc * 32 + 16);
                }
                v16h av = __builtin_shufflevector(lo.v, hi8.v,
                            0,1,2,3,4,5,6,7,8,9,10,11,12,13,14,15);
                v16h b0 = *(const v16h*)(smemB + (size_t)kc * 512 + lane * 16);
                v16h b1 = *(const v16h*)(smemB + (size_t)(nchunks + kc) * 512 + lane * 16);
                acc0 = __builtin_amdgcn_wmma_f32_16x16x32_f16(
                           false, av, false, b0, (short)0, acc0, false, false);
                acc1 = __builtin_amdgcn_wmma_f32_16x16x32_f16(
                           false, av, false, b1, (short)0, acc1, false, false);
            }
        }
    }

    int n0 = (2 * ntPair) * 16 + l15;
    int n1 = n0 + 16;
    if (n0 >= Cout) return;
    float s0 = gamma[n0] * rsqrtf(1.0f + BN_EPS);
    float q0 = beta[n0];
    bool  nv1 = (n1 < Cout);
    float s1 = nv1 ? gamma[n1] * rsqrtf(1.0f + BN_EPS) : 0.f;
    float q1 = nv1 ? beta[n1] : 0.f;
    #pragma unroll
    for (int rI = 0; rI < 8; ++rI) {
        int mm = mbase + rI + 8 * hi;       // D layout: VGPR r -> M = r + 8*laneHi
        if (mm < M) {
            int oy2 = mm / Wo, ox2 = mm - oy2 * Wo;
            _Float16* op = out + (((size_t)b * Ho + oy2) * Wo + ox2) * CdstStride + CoffDst;
            float v0 = fmaxf(acc0[rI] * s0 + q0, 0.0f);
            op[n0] = (_Float16)v0;
            if (nv1) {
                float v1 = fmaxf(acc1[rI] * s1 + q1, 0.0f);
                op[n1] = (_Float16)v1;
            }
        }
    }
}

// ===========================================================================
// Bilinear x2 upsample (reference linspace interp), NHWC, stride == C.
// ===========================================================================
__global__ void upsample_bilinear2_kernel(const _Float16* __restrict__ in,
                                          _Float16* __restrict__ out,
                                          int B, int C, int H, int W)
{
    int Ho = 2 * H, Wo = 2 * W;
    size_t idx = (size_t)blockIdx.x * blockDim.x + threadIdx.x;
    size_t tot = (size_t)B * Ho * Wo * C;
    if (idx >= tot) return;
    int c  = (int)(idx % C); size_t t = idx / C;
    int wo = (int)(t % Wo);  t /= Wo;
    int ho = (int)(t % Ho);  int b = (int)(t / Ho);

    float cy = (float)ho * (float)(H - 1) / (float)(Ho - 1);
    float cx = (float)wo * (float)(W - 1) / (float)(Wo - 1);
    int iy = (int)floorf(cy); iy = iy < 0 ? 0 : (iy > H - 2 ? H - 2 : iy);
    int ix = (int)floorf(cx); ix = ix < 0 ? 0 : (ix > W - 2 ? W - 2 : ix);
    float wy = cy - (float)iy, wx = cx - (float)ix;

    const _Float16* p = in + (size_t)b * H * W * C;
    float v00 = (float)p[((size_t)iy * W + ix) * C + c];
    float v01 = (float)p[((size_t)iy * W + ix + 1) * C + c];
    float v10 = (float)p[((size_t)(iy + 1) * W + ix) * C + c];
    float v11 = (float)p[((size_t)(iy + 1) * W + ix + 1) * C + c];
    float top = v00 + (v01 - v00) * wx;
    float bot = v10 + (v11 - v10) * wx;
    out[idx] = (_Float16)(top + (bot - top) * wy);
}

// ===========================================================================
// Nearest x2 of a 1-ch f32 disparity into an NHWC concat slot (channel Coff);
// also zeroes the pad channels [Coff+1, Cs).
// ===========================================================================
__global__ void nearest_up_disp_kernel(const float* __restrict__ src,
                                       _Float16* __restrict__ dst,
                                       int B, int H, int W,     // src dims
                                       int Cs, int Coff)
{
    int Ho = 2 * H, Wo = 2 * W;
    int idx = blockIdx.x * blockDim.x + threadIdx.x;
    int tot = B * Ho * Wo;
    if (idx >= tot) return;
    int wo = idx % Wo; int t = idx / Wo; int ho = t % Ho; int b = t / Ho;
    float v = src[((size_t)b * H + (ho >> 1)) * W + (wo >> 1)];
    _Float16* p = dst + (((size_t)b * Ho + ho) * Wo + wo) * Cs;
    p[Coff] = (_Float16)v;
    for (int c = Coff + 1; c < Cs; ++c) p[c] = (_Float16)0.f;
}

// ===========================================================================
// Disparity head: 3x3 conv cin->1 + bias, 2*sigmoid, NHWC fp16 in, f32 out.
// ===========================================================================
__global__ void disp_head_kernel(const _Float16* __restrict__ in,   // NHWC stride C
                                 const float* __restrict__ w,       // [1,C,3,3] f32
                                 const float* __restrict__ bias,
                                 float* __restrict__ out,
                                 int B, int C, int H, int W)
{
    int idx = blockIdx.x * blockDim.x + threadIdx.x;
    int HW = H * W;
    if (idx >= B * HW) return;
    int b = idx / HW, r = idx % HW;
    int h = r / W, wo = r % W;

    float acc = 0.0f;
    const _Float16* ib = in + (size_t)b * HW * C;
    #pragma unroll
    for (int ky = 0; ky < 3; ++ky) {
        int y = h + ky - 1;
        if (y < 0 || y >= H) continue;
        #pragma unroll
        for (int kx = 0; kx < 3; ++kx) {
            int x = wo + kx - 1;
            if (x < 0 || x >= W) continue;
            const _Float16* p = ib + ((size_t)y * W + x) * C;
            const float* wr = w + (size_t)ky * 3 + kx;
            for (int c = 0; c < C; ++c)
                acc += (float)p[c] * wr[(size_t)c * 9];
        }
    }
    acc += bias[0];
    out[idx] = 2.0f / (1.0f + expf(-acc));
}

// ===========================================================================
// Host orchestration
// ===========================================================================
extern "C" void kernel_launch(void* const* d_in, const int* in_sizes, int n_in,
                              void* d_out, int out_size, void* d_ws, size_t ws_size,
                              hipStream_t stream)
{
    (void)in_sizes; (void)n_in; (void)out_size; (void)ws_size;
    const int B = 2, H = 256, W = 320;

    const float* left  = (const float*)d_in[0];
    const float* right = (const float*)d_in[1];
    const float* krk   = (const float*)d_in[2];
    const float* kt    = (const float*)d_in[3];
    auto F = [&](int i) -> const float* { return (const float*)d_in[i]; };
    // pytree (sorted-key) leaf order:
    // conv1: b1=4 b2=5 g1=6 g2=7 w1=8 w2=9 ; conv2@10 conv3@16 conv4@22 conv5@28
    // disp1: b=34 w=35 ; disp2@36 disp3@38 disp4@40
    // iconv1: b=42 g=43 w=44 ; iconv2@45 iconv3@48 iconv4@51 iconv5@54
    // upconv1: b=57 g=58 w=59 ; upconv2@60 upconv3@63 upconv4@66 upconv5@69

    float* dout = (float*)d_out;
    float* d1o = dout;                 // 2*256*320
    float* d2o = dout + 163840;        // 2*128*160
    float* d3o = dout + 204800;        // 2*64*80
    float* d4o = dout + 215040;        // 2*32*40

    char* ws = (char*)d_ws;
    size_t off = 0;
    auto alloc16 = [&](size_t elems) -> _Float16* {
        _Float16* p = (_Float16*)(ws + off);
        off += ((elems * sizeof(_Float16) + 255) / 256) * 256;
        return p;
    };
    auto repack = [&](int widx, int Cout, int Cin, int Kk) -> _Float16* {
        int ntiles = (Cout + 15) / 16, nchunks = (Cin + 31) / 32;
        size_t n = (size_t)Kk * Kk * ntiles * nchunks * 512;
        _Float16* d = alloc16(n);
        repack_weight_kernel<<<(unsigned)((n + 255) / 256), 256, 0, stream>>>(
            F(widx), d, Cout, Cin, Kk, ntiles, nchunks);
        return d;
    };
    auto conv = [&](const _Float16* in, const _Float16* wg,
                    const float* g, const float* bt, _Float16* out,
                    int Cin, int CinStride, int CinOff, int Hin, int Win,
                    int Cout, int Ho, int Wo, int CdstStride, int CoffDst,
                    int Kk, int stride, int pad) {
        int M = Ho * Wo;
        dim3 grid((unsigned)((M + 63) / 64), (unsigned)((Cout + 31) / 32), (unsigned)B);
        conv_wmma_kernel<<<grid, 128, 0, stream>>>(in, wg, g, bt, out,
            Cin, CinStride, CinOff, Hin, Win, Cout, Ho, Wo, CdstStride, CoffDst,
            Kk, stride, pad);
    };
    auto up2 = [&](const _Float16* in, _Float16* out, int C, int h, int w) {
        size_t tot = (size_t)B * C * 4 * h * w;
        upsample_bilinear2_kernel<<<(unsigned)((tot + 255) / 256), 256, 0, stream>>>(
            in, out, B, C, h, w);
    };
    auto nearestUp = [&](const float* src, _Float16* dst, int h, int w, int Cs, int Co) {
        int tot = B * 4 * h * w;
        nearest_up_disp_kernel<<<(unsigned)((tot + 255) / 256), 256, 0, stream>>>(
            src, dst, B, h, w, Cs, Co);
    };
    auto disp = [&](const _Float16* in, int widx, int bidx, float* out, int C, int h, int w) {
        int tot = B * h * w;
        disp_head_kernel<<<(unsigned)((tot + 255) / 256), 256, 0, stream>>>(
            in, F(widx), F(bidx), out, B, C, h, w);
    };

    // NHWC feature buffers (channel strides padded to 32 where needed)
    const int HW1 = 81920, HW2 = 20480, HW3 = 5120, HW4 = 1280, HW5 = 320, HW6 = 80;
    _Float16* X     = alloc16((size_t)B * HW1 *   96);  // 67 used
    _Float16* T1    = alloc16((size_t)B * HW1 *  128);
    _Float16* C1CAT = alloc16((size_t)B * HW2 *  288);  // u2|c1|d3 (257 used)
    _Float16* T2    = alloc16((size_t)B * HW2 *  256);
    _Float16* C2CAT = alloc16((size_t)B * HW3 *  544);  // u3|c2|d4 (513 used)
    _Float16* T3    = alloc16((size_t)B * HW3 *  512);
    _Float16* C3CAT = alloc16((size_t)B * HW4 * 1024);  // u4|c3
    _Float16* T4    = alloc16((size_t)B * HW4 *  512);
    _Float16* C4CAT = alloc16((size_t)B * HW5 * 1024);  // u5|c4
    _Float16* T5    = alloc16((size_t)B * HW5 *  512);
    _Float16* C5    = alloc16((size_t)B * HW6 *  512);
    _Float16* UP5   = alloc16((size_t)B * HW5 *  512);
    _Float16* I5    = alloc16((size_t)B * HW5 *  512);
    _Float16* UPI5  = alloc16((size_t)B * HW4 *  512);
    _Float16* I4b   = alloc16((size_t)B * HW4 *  512);
    _Float16* UPI4  = alloc16((size_t)B * HW3 *  512);
    _Float16* I3b   = alloc16((size_t)B * HW3 *  256);
    _Float16* UPI3  = alloc16((size_t)B * HW2 *  256);
    _Float16* I2b   = alloc16((size_t)B * HW2 *  128);
    _Float16* UPI2  = alloc16((size_t)B * HW1 *  128);
    _Float16* C0CAT = alloc16((size_t)B * HW1 *   96);  // u1|d2 (65 used)
    _Float16* I1b   = alloc16((size_t)B * HW1 *   64);

    // fragment-major fp16 weights
    _Float16* w_c1a = repack(8,  128,   67, 7);
    _Float16* w_c1b = repack(9,  128,  128, 7);
    _Float16* w_c2a = repack(14, 256,  128, 5);
    _Float16* w_c2b = repack(15, 256,  256, 5);
    _Float16* w_c3a = repack(20, 512,  256, 3);
    _Float16* w_c3b = repack(21, 512,  512, 3);
    _Float16* w_c4a = repack(26, 512,  512, 3);
    _Float16* w_c4b = repack(27, 512,  512, 3);
    _Float16* w_c5a = repack(32, 512,  512, 3);
    _Float16* w_c5b = repack(33, 512,  512, 3);
    _Float16* w_up5 = repack(71, 512,  512, 3);
    _Float16* w_i5  = repack(56, 512, 1024, 3);
    _Float16* w_up4 = repack(68, 512,  512, 3);
    _Float16* w_i4  = repack(53, 512, 1024, 3);
    _Float16* w_up3 = repack(65, 256,  512, 3);
    _Float16* w_i3  = repack(50, 256,  513, 3);
    _Float16* w_up2 = repack(62, 128,  256, 3);
    _Float16* w_i2  = repack(47, 128,  257, 3);
    _Float16* w_up1 = repack(59,  64,  128, 3);
    _Float16* w_i1  = repack(44,  64,   65, 3);

    // cost volume -> X (also zeroes X pad channels)
    {
        int tot = B * H * W;
        cost_volume_kernel<<<(unsigned)((tot + 255) / 256), 256, 0, stream>>>(
            left, right, krk, kt, X, B, H, W);
    }

    // ---------------- encoder ----------------
    conv(X,     w_c1a, F(6),  F(4),  T1,     67,   96,   0, 256, 320, 128, 256, 320,  128,   0, 7, 1, 3);
    conv(T1,    w_c1b, F(7),  F(5),  C1CAT, 128,  128,   0, 256, 320, 128, 128, 160,  288, 128, 7, 2, 3);
    conv(C1CAT, w_c2a, F(12), F(10), T2,    128,  288, 128, 128, 160, 256, 128, 160,  256,   0, 5, 1, 2);
    conv(T2,    w_c2b, F(13), F(11), C2CAT, 256,  256,   0, 128, 160, 256,  64,  80,  544, 256, 5, 2, 2);
    conv(C2CAT, w_c3a, F(18), F(16), T3,    256,  544, 256,  64,  80, 512,  64,  80,  512,   0, 3, 1, 1);
    conv(T3,    w_c3b, F(19), F(17), C3CAT, 512,  512,   0,  64,  80, 512,  32,  40, 1024, 512, 3, 2, 1);
    conv(C3CAT, w_c4a, F(24), F(22), T4,    512, 1024, 512,  32,  40, 512,  32,  40,  512,   0, 3, 1, 1);
    conv(T4,    w_c4b, F(25), F(23), C4CAT, 512,  512,   0,  32,  40, 512,  16,  20, 1024, 512, 3, 2, 1);
    conv(C4CAT, w_c5a, F(30), F(28), T5,    512, 1024, 512,  16,  20, 512,  16,  20,  512,   0, 3, 1, 1);
    conv(T5,    w_c5b, F(31), F(29), C5,    512,  512,   0,  16,  20, 512,   8,  10,  512,   0, 3, 2, 1);

    // ---------------- decoder ----------------
    up2(C5, UP5, 512, 8, 10);
    conv(UP5,   w_up5, F(70), F(69), C4CAT,  512,  512, 0, 16, 20, 512, 16, 20, 1024, 0, 3, 1, 1); // u5
    conv(C4CAT, w_i5,  F(55), F(54), I5,    1024, 1024, 0, 16, 20, 512, 16, 20,  512, 0, 3, 1, 1);

    up2(I5, UPI5, 512, 16, 20);
    conv(UPI5,  w_up4, F(67), F(66), C3CAT,  512,  512, 0, 32, 40, 512, 32, 40, 1024, 0, 3, 1, 1); // u4
    conv(C3CAT, w_i4,  F(52), F(51), I4b,   1024, 1024, 0, 32, 40, 512, 32, 40,  512, 0, 3, 1, 1);
    disp(I4b, 41, 40, d4o, 512, 32, 40);

    up2(I4b, UPI4, 512, 32, 40);
    conv(UPI4,  w_up3, F(64), F(63), C2CAT,  512,  512, 0, 64, 80, 256, 64, 80,  544, 0, 3, 1, 1); // u3
    nearestUp(d4o, C2CAT, 32, 40, 544, 512);
    conv(C2CAT, w_i3,  F(49), F(48), I3b,    513,  544, 0, 64, 80, 256, 64, 80,  256, 0, 3, 1, 1);
    disp(I3b, 39, 38, d3o, 256, 64, 80);

    up2(I3b, UPI3, 256, 64, 80);
    conv(UPI3,  w_up2, F(61), F(60), C1CAT,  256,  256, 0, 128, 160, 128, 128, 160, 288, 0, 3, 1, 1); // u2
    nearestUp(d3o, C1CAT, 64, 80, 288, 256);
    conv(C1CAT, w_i2,  F(46), F(45), I2b,    257,  288, 0, 128, 160, 128, 128, 160, 128, 0, 3, 1, 1);
    disp(I2b, 37, 36, d2o, 128, 128, 160);

    up2(I2b, UPI2, 128, 128, 160);
    conv(UPI2,  w_up1, F(58), F(57), C0CAT,  128,  128, 0, 256, 320,  64, 256, 320,  96, 0, 3, 1, 1); // u1
    nearestUp(d2o, C0CAT, 128, 160, 96, 64);
    conv(C0CAT, w_i1,  F(43), F(42), I1b,     65,   96, 0, 256, 320,  64, 256, 320,  64, 0, 3, 1, 1);
    disp(I1b, 35, 34, d1o, 64, 256, 320);
}
